// MLA_88399016886295
// MI455X (gfx1250) — matrix operand order repaired and checked by reading, not compiled, not run on hardware
//
#include <hip/hip_runtime.h>

// ---------------------------------------------------------------------------
// Types for CDNA5 WMMA (gfx1250, wave32)
// ---------------------------------------------------------------------------
typedef unsigned short u16;
typedef __attribute__((ext_vector_type(16))) __bf16 v16bf;
typedef __attribute__((ext_vector_type(8)))  __bf16 v8bf;
typedef __attribute__((ext_vector_type(8)))  float  v8f;
typedef __attribute__((ext_vector_type(4)))  unsigned int u32x4;
typedef __attribute__((ext_vector_type(8)))  int i32x8;
typedef __attribute__((ext_vector_type(4)))  int i32x4;

union FragU { v16bf v; v8bf h8[2]; u16 s[16]; };

#if __has_builtin(__builtin_amdgcn_tensor_load_to_lds) && __has_builtin(__builtin_amdgcn_s_wait_tensorcnt)
#define HAVE_TDM 1
#else
#define HAVE_TDM 0
#endif

__device__ __forceinline__ u16 f2bf(float x) {
    unsigned int u = __float_as_uint(x);
    unsigned int r = (u + 0x7FFFu + ((u >> 16) & 1u)) >> 16;
    return (u16)r;
}

__device__ __forceinline__ v8f wmma_bf16(v16bf a, v16bf b, v8f c) {
    return __builtin_amdgcn_wmma_f32_16x16x32_bf16(
        false, a, false, b, (short)0, c, false, false);
}

// A fragment (16x32 bf16): lane<16: M=lane, K=k+0..7 & k+16..23 ; lane>=16: M=lane-16, K=k+8..15 & k+24..31
__device__ __forceinline__ v16bf frag_a_lds(const u16* base, int ld, int m, int k, int lane) {
    int l  = lane & 15;
    int kh = (lane >> 4) << 3;
    const u16* q = base + (m + l) * ld + k + kh;
    FragU fu;
    fu.h8[0] = *(const v8bf*)q;
    fu.h8[1] = *(const v8bf*)(q + 16);
    return fu.v;
}

// B fragment (32x16 bf16): lane<16: N=lane, K=k+0..7 & k+16..23 ; lane>=16: N=lane-16, K=k+8..15 & k+24..31. base stored [N][K].
__device__ __forceinline__ v16bf frag_b_lds(const u16* base, int ld, int n, int k, int lane) {
    int l  = lane & 15;
    int kh = (lane >> 4) << 3;
    const u16* q = base + (n + l) * ld + k + kh;
    FragU fu;
    fu.h8[0] = *(const v8bf*)q;
    fu.h8[1] = *(const v8bf*)(q + 16);
    return fu.v;
}

// A fragment sourced from f32 LDS (converted on the fly) -- used for P in P*V
__device__ __forceinline__ v16bf frag_a_f32(const float* base, int ld, int m, int k, int lane) {
    int l  = lane & 15;
    int kh = (lane >> 4) << 3;
    const float* q = base + (m + l) * ld + k + kh;
    FragU fu;
#pragma unroll
    for (int i = 0; i < 8; ++i) fu.s[i] = f2bf(q[i]);
#pragma unroll
    for (int i = 0; i < 8; ++i) fu.s[8 + i] = f2bf(q[16 + i]);
    return fu.v;
}

// ---------------------------------------------------------------------------
// Tensor Data Mover: 2D tile (rows x width bf16 elems, row stride in elems)
// from global memory into LDS at byte offset lds_off. Issued per-wave.
// ---------------------------------------------------------------------------
#if HAVE_TDM
__device__ __forceinline__ void tdm_load_2d(unsigned lds_off, const u16* gsrc,
                                            int width, int rows, long stride_elems) {
    unsigned long long ga = (unsigned long long)(const void*)gsrc;
    u32x4 g0;
    g0[0] = 1u;                                             // count=1, user descriptor
    g0[1] = lds_off;                                        // lds_addr (bytes)
    g0[2] = (unsigned)(ga & 0xFFFFFFFFull);                 // global_addr[31:0]
    g0[3] = (unsigned)((ga >> 32) & 0x1FFFFFFull) | (2u << 30);  // addr[56:32] | type=2
    unsigned long long st = (unsigned long long)stride_elems;
    i32x8 g1;
    g1[0] = (int)(1u << 16);                                // wg_mask=0, data_size=1 (2B)
    g1[1] = (int)(((unsigned)width & 0xFFFFu) << 16);       // tensor_dim0[15:0]
    g1[2] = (int)((((unsigned)width >> 16) & 0xFFFFu) |
                  (((unsigned)rows  & 0xFFFFu) << 16));     // tensor_dim0_hi | tensor_dim1_lo
    g1[3] = (int)((((unsigned)rows >> 16) & 0xFFFFu) |
                  (((unsigned)width & 0xFFFFu) << 16));     // tensor_dim1_hi | tile_dim0
    g1[4] = (int)((unsigned)rows & 0xFFFFu);                // tile_dim1 | tile_dim2=0
    g1[5] = (int)(st & 0xFFFFFFFFull);                      // tensor_dim0_stride[31:0]
    g1[6] = (int)((st >> 32) & 0xFFFFull);                  // stride[47:32] | dim1_stride_lo=0
    g1[7] = 0;
    i32x4 z = {0, 0, 0, 0};
#if defined(__clang_major__) && (__clang_major__ >= 23)
    i32x8 z8 = {0, 0, 0, 0, 0, 0, 0, 0};
    __builtin_amdgcn_tensor_load_to_lds(g0, g1, z, z, z8, 0);
#else
    __builtin_amdgcn_tensor_load_to_lds(g0, g1, z, z, 0);
#endif
}
#endif

// ---------------------------------------------------------------------------
// Elementwise kernels (4-wide)
// ---------------------------------------------------------------------------
__global__ void f32_to_bf16_kernel(const float* __restrict__ in, u16* __restrict__ out, long n4) {
    long i = (long)blockIdx.x * blockDim.x + threadIdx.x;
    if (i < n4) {
        float4 f = ((const float4*)in)[i];
        union { ushort4 u4; u16 s[4]; } r;
        r.s[0] = f2bf(f.x); r.s[1] = f2bf(f.y); r.s[2] = f2bf(f.z); r.s[3] = f2bf(f.w);
        ((ushort4*)out)[i] = r.u4;
    }
}

// RoPE on (B,S,Hn,128) fp32 -> bf16, one thread per (b,s,h,d<64) pair.
__global__ void rope_bf16_kernel(const float* __restrict__ in, u16* __restrict__ out,
                                 int Sn, int Hn, const int* __restrict__ start_pos, long total) {
    long idx = (long)blockIdx.x * blockDim.x + threadIdx.x;
    if (idx >= total) return;
    int d = (int)(idx & 63);
    long t = idx >> 6;
    int h = (int)(t % Hn); t /= Hn;
    int s = (int)(t % Sn);
    long b = t / Sn;
    float pos = (float)(s + start_pos[0]);
    float ang = pos * __expf(-((float)d) * (9.210340371976184f / 64.0f));   // 10000^(-d/64)
    float c = __cosf(ang), sn = __sinf(ang);
    long base = ((b * Sn + s) * (long)Hn + h) * 128;
    float a  = in[base + d];
    float bb = in[base + d + 64];
    out[base + d]      = f2bf(a * c - bb * sn);
    out[base + d + 64] = f2bf(bb * c + a * sn);
}

// combined = attn_out(B,S,2048) + tile(latent_out(B,S,128), 16)  -> bf16 (4-wide)
__global__ void combine_kernel(const float* __restrict__ attn, const float* __restrict__ lat,
                               u16* __restrict__ outb, long n4) {
    long i = (long)blockIdx.x * blockDim.x + threadIdx.x;
    if (i < n4) {
        long e = i << 2;
        long row = e >> 11;
        int  col = (int)(e & 2047);
        float4 a = ((const float4*)attn)[i];
        float4 l = *(const float4*)&lat[row * 128 + (col & 127)];
        union { ushort4 u4; u16 s[4]; } r;
        r.s[0] = f2bf(a.x + l.x);
        r.s[1] = f2bf(a.y + l.y);
        r.s[2] = f2bf(a.z + l.z);
        r.s[3] = f2bf(a.w + l.w);
        ((ushort4*)outb)[i] = r.u4;
    }
}

// ---------------------------------------------------------------------------
// bf16 WMMA GEMM:  C[M,N] = A[M,K] * B[K,N] (both row major)
// Block 128x128, BK=32, 256 threads = 8 waves (2M x 4N), wave tile 64x32.
// OUT_BF16 templated -> branch-free epilogue.
// ---------------------------------------------------------------------------
template <int OUT_BF16>
__global__ __launch_bounds__(256) void gemm_bf16_kernel(
    const u16* __restrict__ A, const u16* __restrict__ Bm,
    void* __restrict__ Cout, int M, int N, int K)
{
    constexpr int GBK = 32;
    __shared__ __attribute__((aligned(32))) u16 As[128 * GBK];   // [m][k]
    __shared__ __attribute__((aligned(32))) u16 Bs[128 * GBK];   // [n][k]

    int tid  = threadIdx.x;
    int lane = tid & 31, wid = tid >> 5;
    int wm = wid >> 2, wn = wid & 3;
    int laneHi = lane >> 4;
    int m0 = blockIdx.x * 128, n0 = blockIdx.y * 128;

    int ra  = tid >> 1;               // A-load row
    int hca = (tid & 1) * 16;         // A-load col chunk
    int kkb = tid >> 3;               // B-load k row
    int ncb = (tid & 7) * 16;         // B-load n chunk

    v8f acc[4][2];
#pragma unroll
    for (int mi = 0; mi < 4; ++mi)
#pragma unroll
        for (int ni = 0; ni < 2; ++ni)
#pragma unroll
            for (int j = 0; j < 8; ++j) acc[mi][ni][j] = 0.0f;

    for (int kb = 0; kb < K; kb += GBK) {
        {   // A tile 128x32 row-major
            const u16* src = &A[(long)(m0 + ra) * K + kb + hca];
            *(v8bf*)&As[ra * GBK + hca]     = *(const v8bf*)src;
            *(v8bf*)&As[ra * GBK + hca + 8] = *(const v8bf*)(src + 8);
        }
        {   // B tile 32x128, transposed into Bs[n][k]
            const u16* src = &Bm[(long)(kb + kkb) * N + n0 + ncb];
            v8bf b0 = *(const v8bf*)src;
            v8bf b1 = *(const v8bf*)(src + 8);
            FragU u0, u1; u0.h8[0] = b0; u1.h8[0] = b1;
#pragma unroll
            for (int i = 0; i < 8; ++i) Bs[(ncb + i) * GBK + kkb]     = u0.s[i];
#pragma unroll
            for (int i = 0; i < 8; ++i) Bs[(ncb + 8 + i) * GBK + kkb] = u1.s[i];
        }
        if (kb + GBK < K) {   // prefetch next tiles (global_prefetch_b8)
            __builtin_prefetch(&A[(long)(m0 + ra) * K + kb + GBK + hca], 0, 1);
            __builtin_prefetch(&Bm[(long)(kb + GBK + kkb) * N + n0 + ncb], 0, 1);
        }
        __syncthreads();

        v16bf bfr[2];
#pragma unroll
        for (int ni = 0; ni < 2; ++ni)
            bfr[ni] = frag_b_lds(Bs, GBK, wn * 32 + ni * 16, 0, lane);
#pragma unroll
        for (int mi = 0; mi < 4; ++mi) {
            v16bf afr = frag_a_lds(As, GBK, wm * 64 + mi * 16, 0, lane);
#pragma unroll
            for (int ni = 0; ni < 2; ++ni)
                acc[mi][ni] = wmma_bf16(afr, bfr[ni], acc[mi][ni]);
        }
        __syncthreads();
    }

#pragma unroll
    for (int mi = 0; mi < 4; ++mi)
#pragma unroll
        for (int ni = 0; ni < 2; ++ni)
#pragma unroll
            for (int j = 0; j < 8; ++j) {
                int r = m0 + wm * 64 + mi * 16 + j + 8 * laneHi;
                int c = n0 + wn * 32 + ni * 16 + (lane & 15);
                float v = acc[mi][ni][j];
                if (OUT_BF16) ((u16*)Cout)[(long)r * N + c]   = f2bf(v);
                else          ((float*)Cout)[(long)r * N + c] = v;
            }
}

// ---------------------------------------------------------------------------
// Flash attention (causal), hd = 128. Q-tile 64, K-tile 128, 8 waves.
// q: (B,S,Hq,128) bf16 ; k,v: (B,S,Hkv,128) bf16 ; out: (B,S,Hq*128) f32
// Q and K tiles are staged into LDS by the Tensor Data Mover when available.
// ---------------------------------------------------------------------------
__global__ __launch_bounds__(256) void flash_attn_kernel(
    const u16* __restrict__ q, const u16* __restrict__ k, const u16* __restrict__ v,
    float* __restrict__ out, int Sn, int Hq, int Hkv, float scale)
{
    constexpr int QT = 64, KT = 128, HD2 = 128;
    __shared__ __attribute__((aligned(32))) u16   Qs[QT * HD2];   // 16 KB
    __shared__ __attribute__((aligned(32))) u16   Ks[KT * HD2];   // 32 KB [n][k]
    __shared__ __attribute__((aligned(32))) u16   Vt[HD2 * KT];   // 32 KB [hd][kk]
    __shared__ __attribute__((aligned(32))) float Ss[QT * KT];    // 32 KB
    __shared__ float mrow[QT], lrow[QT], arow[QT];

    int qt = blockIdx.x, h = blockIdx.y, b = blockIdx.z;
    int hk = h / (Hq / Hkv);
    int tid = threadIdx.x, lane = tid & 31, wid = tid >> 5;
    int wm = wid >> 1;     // 0..3 -> 16-row strip
    int wn = wid & 1;      // 0..1 -> 64-col strip
    int laneHi = lane >> 4;

    const u16* qbase = q + (((long)b * Sn + qt * QT) * Hq + h) * HD2;
#if HAVE_TDM
    if (wid == 0) {
        tdm_load_2d((unsigned)(unsigned long long)(void*)Qs, qbase, HD2, QT, (long)Hq * HD2);
        __builtin_amdgcn_s_wait_tensorcnt(0);
    }
#else
    for (int i = tid * 8; i < QT * HD2; i += 256 * 8) {
        int r = i / HD2, c = i % HD2;
        *(v8bf*)&Qs[r * HD2 + c] = *(const v8bf*)&qbase[(long)r * Hq * HD2 + c];
    }
#endif
    if (tid < QT) { mrow[tid] = -3.0e38f; lrow[tid] = 0.0f; }

    v8f oacc[4];
#pragma unroll
    for (int ni = 0; ni < 4; ++ni)
#pragma unroll
        for (int j = 0; j < 8; ++j) oacc[ni][j] = 0.0f;

    __syncthreads();

    int ktiles = (qt * QT + QT + KT - 1) / KT;
    for (int kt = 0; kt < ktiles; ++kt) {
        const u16* kbase = k + (((long)b * Sn + kt * KT) * Hkv + hk) * HD2;
        const u16* vbase = v + (((long)b * Sn + kt * KT) * Hkv + hk) * HD2;
#if HAVE_TDM
        if (wid == 0)   // async K-tile DMA, overlapped with V transpose below
            tdm_load_2d((unsigned)(unsigned long long)(void*)Ks, kbase, HD2, KT, (long)Hkv * HD2);
#else
        for (int i = tid * 8; i < KT * HD2; i += 256 * 8) {
            int r = i / HD2, c = i % HD2;
            *(v8bf*)&Ks[r * HD2 + c] = *(const v8bf*)&kbase[(long)r * Hkv * HD2 + c];
        }
#endif
        // V tile transposed -> Vt[hd][kk]
        for (int i = tid; i < KT * HD2; i += 256) {
            int r = i / HD2, c = i % HD2;
            Vt[c * KT + r] = vbase[(long)r * Hkv * HD2 + c];
        }
#if HAVE_TDM
        if (wid == 0) __builtin_amdgcn_s_wait_tensorcnt(0);
#endif
        __syncthreads();

        // S = Q * K^T (wave: 16 rows x 64 cols = 4 tiles)
        v8f sacc[4];
#pragma unroll
        for (int sn = 0; sn < 4; ++sn)
#pragma unroll
            for (int j = 0; j < 8; ++j) sacc[sn][j] = 0.0f;

        for (int ks = 0; ks < HD2 / 32; ++ks) {
            v16bf afr = frag_a_lds(Qs, HD2, wm * 16, ks * 32, lane);
#pragma unroll
            for (int sn = 0; sn < 4; ++sn) {
                v16bf bfr = frag_b_lds(Ks, HD2, wn * 64 + sn * 16, ks * 32, lane);
                sacc[sn] = wmma_bf16(afr, bfr, sacc[sn]);
            }
        }
        // Scale + causal mask + spill to LDS
#pragma unroll
        for (int sn = 0; sn < 4; ++sn)
#pragma unroll
            for (int j = 0; j < 8; ++j) {
                int mloc = wm * 16 + j + 8 * laneHi;
                int nloc = wn * 64 + sn * 16 + (lane & 15);
                float sv = sacc[sn][j] * scale;
                if (kt * KT + nloc > qt * QT + mloc) sv = -3.0e38f;
                Ss[mloc * KT + nloc] = sv;
            }
        __syncthreads();

        // Online softmax (one thread per row)
        if (tid < QT) {
            float mold = mrow[tid];
            float mx = mold;
            float* srow = &Ss[tid * KT];
            for (int c = 0; c < KT; ++c) mx = fmaxf(mx, srow[c]);
            float alpha = __expf(mold - mx);
            float sum = 0.0f;
            for (int c = 0; c < KT; ++c) {
                float pv = __expf(srow[c] - mx);
                srow[c] = pv;
                sum += pv;
            }
            mrow[tid] = mx;
            lrow[tid] = lrow[tid] * alpha + sum;
            arow[tid] = alpha;
        }
        __syncthreads();

        // Rescale running O and accumulate O += P * V
#pragma unroll
        for (int ni = 0; ni < 4; ++ni)
#pragma unroll
            for (int j = 0; j < 8; ++j)
                oacc[ni][j] *= arow[wm * 16 + j + 8 * laneHi];

        for (int ks = 0; ks < KT / 32; ++ks) {
            v16bf pfr = frag_a_f32(Ss, KT, wm * 16, ks * 32, lane);
#pragma unroll
            for (int ni = 0; ni < 4; ++ni) {
                v16bf vfr = frag_b_lds(Vt, KT, wn * 64 + ni * 16, ks * 32, lane);
                oacc[ni] = wmma_bf16(pfr, vfr, oacc[ni]);
            }
        }
        __syncthreads();
    }

    // Epilogue: normalize by l and store (B,S,Hq*128)
    float* obase = out + ((long)b * Sn + qt * QT) * (long)(Hq * HD2) + (long)h * HD2;
#pragma unroll
    for (int ni = 0; ni < 4; ++ni)
#pragma unroll
        for (int j = 0; j < 8; ++j) {
            int r = wm * 16 + j + 8 * laneHi;
            int c = wn * 64 + ni * 16 + (lane & 15);
            obase[(long)r * Hq * HD2 + c] = oacc[ni][j] * __frcp_rn(lrow[r]);
        }
}

// ---------------------------------------------------------------------------
// Host-side orchestration
// ---------------------------------------------------------------------------
extern "C" void kernel_launch(void* const* d_in, const int* in_sizes, int n_in,
                              void* d_out, int out_size, void* d_ws, size_t ws_size,
                              hipStream_t stream) {
    (void)in_sizes; (void)n_in; (void)out_size; (void)ws_size;

    const float* x    = (const float*)d_in[0];
    /* d_in[1] = mask (causal) -- recomputed on device */
    const float* wq   = (const float*)d_in[2];
    const float* wk   = (const float*)d_in[3];
    const float* wv   = (const float*)d_in[4];
    const float* wql  = (const float*)d_in[5];
    const float* wkl  = (const float*)d_in[6];
    const float* wo   = (const float*)d_in[7];
    const int*   spos = (const int*)d_in[8];
    float* out = (float*)d_out;

    const int Bn = 2, Sn = 2048, Dn = 2048, Hn = 16, KVn = 4, HDn = 128, Ln = 128;
    const long MR = (long)Bn * Sn;                 // 4096 token rows
    const float scale = 0.08838834764831845f;      // 128^-0.5 (both HD and L)

    char* p = (char*)d_ws;
    auto alloc = [&](size_t bytes) -> char* {
        char* r = p;
        p += (bytes + 255) & ~(size_t)255;
        return r;
    };

    u16* xb   = (u16*)alloc(MR * Dn * 2);
    u16* wqb  = (u16*)alloc((size_t)Dn * Hn * HDn * 2);
    u16* wkb  = (u16*)alloc((size_t)Dn * KVn * HDn * 2);
    u16* wvb  = (u16*)alloc((size_t)Dn * KVn * HDn * 2);
    u16* wqlb = (u16*)alloc((size_t)Dn * Ln * 2);
    u16* wklb = (u16*)alloc((size_t)Dn * Ln * 2);
    u16* wob  = (u16*)alloc((size_t)Hn * HDn * Dn * 2);
    float* qf = (float*)alloc(MR * Hn * HDn * 4);
    float* kf = (float*)alloc(MR * KVn * HDn * 4);
    u16* qb   = (u16*)alloc(MR * Hn * HDn * 2);
    u16* kb   = (u16*)alloc(MR * KVn * HDn * 2);
    u16* vb   = (u16*)alloc(MR * KVn * HDn * 2);
    u16* qlb  = (u16*)alloc(MR * Ln * 2);
    u16* klb  = (u16*)alloc(MR * Ln * 2);
    float* attn = (float*)alloc(MR * Hn * HDn * 4);
    float* lat  = (float*)alloc(MR * Ln * 4);
    u16* comb   = (u16*)alloc(MR * Hn * HDn * 2);

    auto cvt = [&](const float* src, u16* dst, long n) {
        long n4 = n >> 2;
        f32_to_bf16_kernel<<<dim3((unsigned)((n4 + 255) / 256)), dim3(256), 0, stream>>>(src, dst, n4);
    };

    // 1) bf16 conversions
    cvt(x,   xb,   MR * Dn);
    cvt(wq,  wqb,  (long)Dn * Hn * HDn);
    cvt(wk,  wkb,  (long)Dn * KVn * HDn);
    cvt(wv,  wvb,  (long)Dn * KVn * HDn);
    cvt(wql, wqlb, (long)Dn * Ln);
    cvt(wkl, wklb, (long)Dn * Ln);
    cvt(wo,  wob,  (long)Hn * HDn * Dn);

    // 2) projections (WMMA GEMMs).  v / latents write bf16 directly.
    dim3 blk(256);
    gemm_bf16_kernel<0><<<dim3((unsigned)(MR / 128), (Hn * HDn) / 128), blk, 0, stream>>>(
        xb, wqb, (void*)qf, (int)MR, Hn * HDn, Dn);
    gemm_bf16_kernel<0><<<dim3((unsigned)(MR / 128), (KVn * HDn) / 128), blk, 0, stream>>>(
        xb, wkb, (void*)kf, (int)MR, KVn * HDn, Dn);
    gemm_bf16_kernel<1><<<dim3((unsigned)(MR / 128), (KVn * HDn) / 128), blk, 0, stream>>>(
        xb, wvb, (void*)vb, (int)MR, KVn * HDn, Dn);
    gemm_bf16_kernel<1><<<dim3((unsigned)(MR / 128), Ln / 128), blk, 0, stream>>>(
        xb, wqlb, (void*)qlb, (int)MR, Ln, Dn);
    gemm_bf16_kernel<1><<<dim3((unsigned)(MR / 128), Ln / 128), blk, 0, stream>>>(
        xb, wklb, (void*)klb, (int)MR, Ln, Dn);

    // 3) RoPE on q,k -> bf16
    {
        long tq = (long)Bn * Sn * Hn * 64;
        rope_bf16_kernel<<<dim3((unsigned)((tq + 255) / 256)), dim3(256), 0, stream>>>(
            qf, qb, Sn, Hn, spos, tq);
        long tk = (long)Bn * Sn * KVn * 64;
        rope_bf16_kernel<<<dim3((unsigned)((tk + 255) / 256)), dim3(256), 0, stream>>>(
            kf, kb, Sn, KVn, spos, tk);
    }

    // 4) attention: GQA heads + latent single-head (V = kv_latent)
    flash_attn_kernel<<<dim3(Sn / 64, Hn, Bn), blk, 0, stream>>>(
        qb, kb, vb, attn, Sn, Hn, KVn, scale);
    flash_attn_kernel<<<dim3(Sn / 64, 1, Bn), blk, 0, stream>>>(
        qlb, klb, klb, lat, Sn, 1, 1, scale);

    // 5) combined = attn + tile(latent), as bf16
    {
        long n4 = (MR * (long)(Hn * HDn)) >> 2;
        combine_kernel<<<dim3((unsigned)((n4 + 255) / 256)), dim3(256), 0, stream>>>(
            attn, lat, comb, n4);
    }

    // 6) output projection: out = combined @ wo   (fp32 result)
    gemm_bf16_kernel<0><<<dim3((unsigned)(MR / 128), Dn / 128), blk, 0, stream>>>(
        comb, wob, out, (int)MR, Dn, Hn * HDn);
}
